// RN_70841190580226
// MI455X (gfx1250) — matrix operand order repaired and checked by reading, not compiled, not run on hardware
//
#include <hip/hip_runtime.h>
#include <stdint.h>

#define D 256
#define BATCH 16
#define S 128
#define NTOK (BATCH*S)     // 2048 tokens
#define KB 8               // K blocks of 32 (D/32)
#define NB (D/16)          // 16 N-tiles

typedef __attribute__((ext_vector_type(16))) _Float16 v16h;
typedef __attribute__((ext_vector_type(8)))  float    v8f;

union Frag { uint32_t u[8]; uint4 q[2]; v16h h; };
union Acc  { v8f v; float f[8]; };

__device__ __forceinline__ uint32_t pack2h(float a, float b) {
    union { _Float16 h[2]; uint32_t u; } p;
    p.h[0] = (_Float16)a; p.h[1] = (_Float16)b;
    return p.u;
}

// A-matrix (16x32 f16) per-lane K mapping (ISA 7.12.2):
// lanes 0-15: v0..3 -> K 0..7, v4..7 -> K 16..23 ; lanes 16-31: K 8..15 / 24..31
__device__ __forceinline__ int kmapA(int v, int half) {
    return (v >> 2) * 16 + half * 8 + (v & 3) * 2;
}

// ---- Kernel 1: pack weights (Wl, Wr, Wrel) into B-fragment layout ----
__global__ void k_pack_w(const float* __restrict__ Wl, const float* __restrict__ Wr,
                         const float* __restrict__ Wrel, uint32_t* __restrict__ wpack) {
    int t = blockIdx.x * blockDim.x + threadIdx.x;   // < 3*8*16*32*8 = 98304
    int v    = t & 7;
    int lane = (t >> 3) & 31;
    int nb   = (t >> 8) & 15;
    int kb   = (t >> 12) & 7;
    int mat  = t >> 15;
    const float* W = (mat == 0) ? Wl : ((mat == 1) ? Wr : Wrel);
    int n = nb * 16 + (lane & 15);
    int k = kb * 32 + (v >> 2) * 16 + (lane >> 4) * 8 + (v & 3) * 2;
    wpack[t] = pack2h(W[n * D + k], W[n * D + k + 1]);
}

// ---- Kernel 2: gather emb[X], convert f16, scatter into A-fragment layout ----
__global__ void k_gather(const int* __restrict__ X, const float* __restrict__ emb,
                         uint32_t* __restrict__ apack) {
    int n = blockIdx.x;         // token 0..2047
    int p = threadIdx.x;        // k-pair 0..127
    int x = X[n];
    const float* row = emb + (size_t)x * D;
    int k = p * 2;
    uint32_t u = pack2h(row[k], row[k + 1]);
    int m = n & 15, mb = n >> 4;
    int kb = k >> 5, kk = k & 31;
    int half = (kk >> 3) & 1;
    int v = (kk >> 4) * 4 + ((kk & 7) >> 1);
    apack[((mb * KB + kb) * 32 + (half * 16 + m)) * 8 + v] = u;
}

// ---- Kernel 3: fused L/R projection GEMMs via WMMA ----
__global__ void k_proj(const uint32_t* __restrict__ apack, const uint32_t* __restrict__ wpack,
                       const float* __restrict__ bl, const float* __restrict__ br,
                       float* __restrict__ Lb, float* __restrict__ Rb) {
    int wave = blockIdx.x * (blockDim.x >> 5) + (threadIdx.x >> 5);  // 0..4095
    int lane = threadIdx.x & 31;
    int mat = wave >> 11;            // 0 = L, 1 = R
    int t   = wave & 2047;
    int mb = t >> 4, nb = t & 15;
    v8f acc = {};
    for (int kb = 0; kb < KB; ++kb) {
        Frag a, b;
        const uint4* pa = (const uint4*)(apack + (((mb * KB + kb) * 32 + lane) << 3));
        a.q[0] = pa[0]; a.q[1] = pa[1];
        const uint4* pb = (const uint4*)(wpack + ((((mat * KB + kb) * NB + nb) * 32 + lane) << 3));
        b.q[0] = pb[0]; b.q[1] = pb[1];
        acc = __builtin_amdgcn_wmma_f32_16x16x32_f16(false, a.h, false, b.h,
                                                     (short)0, acc, false, false);
    }
    Acc r; r.v = acc;
    int col = lane & 15, half = lane >> 4;
    const float* bias = mat ? br : bl;
    float bv = bias[nb * 16 + col];
    float* out = mat ? Rb : Lb;
    #pragma unroll
    for (int v2 = 0; v2 < 8; ++v2) {
        int rowi = mb * 16 + half * 8 + v2;
        out[rowi * D + nb * 16 + col] = r.f[v2] + bv;
    }
}

// ---- Kernel 4: pairwise |L_i + R_j| partial sums. ----
// relu(x) = 0.5*(x + |x|): only the |.| term needs the S^2 loop.
// j tiled by 32 into registers -> inner loop is pure 2 VALU/pair
// (v_add + v_add with |src| modifier), 4 accumulators break the FP chain.
__global__ void k_pair_abs(const float* __restrict__ Lb, const float* __restrict__ Rb,
                           float* __restrict__ partial) {
    __shared__ float Rt[64 * D];                 // 64 KB half-tile of R[b]
    int b = blockIdx.x >> 3, it = blockIdx.x & 7;
    int d = threadIdx.x;
    const float* Lbase = Lb + (size_t)b * S * D + (size_t)it * 16 * D;
    const float* Rbase = Rb + (size_t)b * S * D;
    float a0 = 0.f, a1 = 0.f, a2 = 0.f, a3 = 0.f;
    for (int jt = 0; jt < 2; ++jt) {
        const float4* src = (const float4*)(Rbase + jt * 64 * D);
        float4* dst = (float4*)Rt;
        #pragma unroll
        for (int q = 0; q < 16; ++q) dst[d + 256 * q] = src[d + 256 * q];
        __syncthreads();
        for (int jc = 0; jc < 2; ++jc) {
            float r[32];
            #pragma unroll
            for (int jj = 0; jj < 32; ++jj) r[jj] = Rt[(jc * 32 + jj) * D + d];
            for (int i = 0; i < 16; ++i) {
                float l = Lbase[i * D + d];
                #pragma unroll
                for (int jj = 0; jj < 32; jj += 4) {
                    a0 += __builtin_fabsf(l + r[jj + 0]);
                    a1 += __builtin_fabsf(l + r[jj + 1]);
                    a2 += __builtin_fabsf(l + r[jj + 2]);
                    a3 += __builtin_fabsf(l + r[jj + 3]);
                }
            }
        }
        __syncthreads();
    }
    partial[blockIdx.x * D + d] = (a0 + a1) + (a2 + a3);
}

// ---- Kernel 5: deterministic reduce of partials + analytic linear term ----
// pooled[b,d] = ( 0.5*(S*(sumL+sumR) + sum|L_i+R_j|) ) / S^2
__global__ void k_reduce(const float* __restrict__ Lb, const float* __restrict__ Rb,
                         const float* __restrict__ partial, float* __restrict__ pooled) {
    int b = blockIdx.x, d = threadIdx.x;
    float absum = 0.f;
    #pragma unroll
    for (int it = 0; it < 8; ++it) absum += partial[(b * 8 + it) * D + d];
    float sl = 0.f, sr = 0.f;
    const float* Lbase = Lb + (size_t)b * S * D;
    const float* Rbase = Rb + (size_t)b * S * D;
    for (int i = 0; i < S; ++i) { sl += Lbase[i * D + d]; sr += Rbase[i * D + d]; }
    pooled[b * D + d] = 0.5f * ((float)S * (sl + sr) + absum) * (1.f / ((float)S * (float)S));
}

// ---- Kernel 6: final GEMM out = pooled * Wrel^T + brel via WMMA (M=16, one tile) ----
__global__ void k_final(const uint32_t* __restrict__ wpack, const float* __restrict__ pooled,
                        const float* __restrict__ brel, float* __restrict__ out) {
    int nb   = threadIdx.x >> 5;    // 16 waves, one N-tile each
    int lane = threadIdx.x & 31;
    int m = lane & 15, half = lane >> 4;
    v8f acc = {};
    for (int kb = 0; kb < KB; ++kb) {
        Frag a, b;
        #pragma unroll
        for (int v2 = 0; v2 < 8; ++v2) {
            int k = kb * 32 + kmapA(v2, half);
            a.u[v2] = pack2h(pooled[m * D + k], pooled[m * D + k + 1]);
        }
        const uint4* pb = (const uint4*)(wpack + ((((2 * KB + kb) * NB + nb) * 32 + lane) << 3));
        b.q[0] = pb[0]; b.q[1] = pb[1];
        acc = __builtin_amdgcn_wmma_f32_16x16x32_f16(false, a.h, false, b.h,
                                                     (short)0, acc, false, false);
    }
    Acc r; r.v = acc;
    int col = lane & 15;
    float bv = brel[nb * 16 + col];
    #pragma unroll
    for (int v2 = 0; v2 < 8; ++v2)
        out[(half * 8 + v2) * D + nb * 16 + col] = r.f[v2] + bv;
}

extern "C" void kernel_launch(void* const* d_in, const int* in_sizes, int n_in,
                              void* d_out, int out_size, void* d_ws, size_t ws_size,
                              hipStream_t stream) {
    const int*   X    = (const int*)d_in[0];
    const float* emb  = (const float*)d_in[1];
    const float* Wl   = (const float*)d_in[2];
    const float* bl   = (const float*)d_in[3];
    const float* Wr   = (const float*)d_in[4];
    const float* br   = (const float*)d_in[5];
    const float* Wrel = (const float*)d_in[6];
    const float* brel = (const float*)d_in[7];
    float* out = (float*)d_out;

    char* ws = (char*)d_ws;
    uint32_t* wpack   = (uint32_t*)(ws);                 //  384 KB packed weights
    uint32_t* apack   = (uint32_t*)(ws + 393216);        // 1024 KB packed embeddings
    float*    Lb      = (float*)(ws + 1441792);          // 2 MB
    float*    Rb      = (float*)(ws + 3538944);          // 2 MB
    float*    partial = (float*)(ws + 5636096);          // 128 KB  (128 x 256)
    float*    pooled  = (float*)(ws + 5767168);          // 16 KB

    k_pack_w  <<<384, 256, 0, stream>>>(Wl, Wr, Wrel, wpack);
    k_gather  <<<NTOK, 128, 0, stream>>>(X, emb, apack);
    k_proj    <<<512, 256, 0, stream>>>(apack, wpack, bl, br, Lb, Rb);
    k_pair_abs<<<BATCH * 8, 256, 0, stream>>>(Lb, Rb, partial);
    k_reduce  <<<BATCH, 256, 0, stream>>>(Lb, Rb, partial, pooled);
    k_final   <<<1, 512, 0, stream>>>(wpack, pooled, brel, out);
}
